// GM_GAT_81028853006977
// MI455X (gfx1250) — matrix operand order repaired and checked by reading, not compiled, not run on hardware
//
#include <hip/hip_runtime.h>
#include <hip/hip_bf16.h>

// ---------------- problem constants (match reference) ----------------
#define N_NODES 50000            // == 3125 * 16, == 625 * 80
#define N_EDGES 400000
#define ET (N_EDGES + N_NODES)   // edges + self loops = 450000
#define IN_DIM 128
#define HID 64
#define HEADS 3
#define HC (HEADS * HID)         // 192
#define NCLS 40
#define NCLS_PAD 48              // padded to 3 full 16-col tiles
#define NEG_SLOPE 0.2f
#define MT 5                     // M-tiles (of 16 rows) per wave in the GEMM

typedef __attribute__((ext_vector_type(16))) _Float16 v16h;
typedef __attribute__((ext_vector_type(8)))  _Float16 v8h;
typedef __attribute__((ext_vector_type(4)))  _Float16 v4h;
typedef __attribute__((ext_vector_type(8)))  float    v8f;

// ---------------- helpers ----------------
__device__ __forceinline__ unsigned fenc(float v) {
  unsigned u = __float_as_uint(v);
  return (u & 0x80000000u) ? ~u : (u | 0x80000000u);  // order-preserving map
}
__device__ __forceinline__ float fdec(unsigned u) {
  u = (u & 0x80000000u) ? (u & 0x7FFFFFFFu) : ~u;
  return __uint_as_float(u);
}

// ---------------- f32 -> f16 convert, 4-wide ----------------
__global__ void cvt_f32_f16x4(const float4* __restrict__ src, v4h* __restrict__ dst, int n4) {
  int i = blockIdx.x * blockDim.x + threadIdx.x;
  if (i >= n4) return;
  float4 v = src[i];
  v4h o;
  o[0] = (_Float16)v.x; o[1] = (_Float16)v.y; o[2] = (_Float16)v.z; o[3] = (_Float16)v.w;
  dst[i] = o;
}

// ---------------- WMMA GEMM: C[M,ldC] = A[M,K](f16) * W[OCpad,K](f16)^T (+bias) ----------------
// K compile-time (128 or 192) -> fully unrolled, double-buffered fragments so
// the loads of k-step i+1 overlap the 5 WMMAs of k-step i (partial loadcnt waits).
// Preconditions baked in: M % (16*MT) == 0, K % 32 == 0, OCpad % 16 == 0.
// block = 128 threads = 4 waves; wave w owns 5 stacked 16x16 tiles at
// rows [blockIdx.x*80, +80), cols [(blockIdx.y*4+w)*16, +16). EXEC stays all-1.
template <int K>
__global__ __launch_bounds__(128)
void gemm_wmma_f16(const _Float16* __restrict__ A, const _Float16* __restrict__ W,
                   const float* __restrict__ bias, float* __restrict__ C,
                   int OCpad, int ldC) {
  const int lane = threadIdx.x & 31;
  const int wave = threadIdx.x >> 5;
  const int m0 = blockIdx.x * (16 * MT);
  const int n0 = (blockIdx.y * 4 + wave) * 16;
  if (n0 >= OCpad) return;                 // wave-uniform
  const int half = lane >> 4;              // 0: lanes 0-15, 1: lanes 16-31
  const int l15  = lane & 15;

  // A fragment (16x32 f16): lane = row m0+l15 (+16 per tile), K pairs:
  //   lanes 0-15 : K = k0+0..7 and k0+16..23 ; lanes 16-31: K = k0+8..15 and k0+24..31
  const _Float16* Arow = A + (size_t)(m0 + l15) * K + (half << 3);
  const size_t tstride = (size_t)16 * K;   // one M-tile of rows

  // B fragment (32x16 f16): lane = column n0+l15 (= row of W),
  //   lanes 0-15: K=k0+0..7 and k0+16..23, lanes 16-31: K=k0+8..15 and k0+24..31
  const _Float16* Wrow = W + (size_t)(n0 + l15) * K + (half << 3);

  auto loadB = [&](int k0, v16h& b) {
    v8h lo = *(const v8h*)(Wrow + k0);
    v8h hi = *(const v8h*)(Wrow + k0 + 16);
#pragma unroll
    for (int i = 0; i < 8; ++i) { b[i] = lo[i]; b[i + 8] = hi[i]; }
  };
  auto loadA = [&](int k0, v16h* a) {
#pragma unroll
    for (int t = 0; t < MT; ++t) {
      const _Float16* ap = Arow + t * tstride + k0;
      v8h lo = *(const v8h*)(ap);
      v8h hi = *(const v8h*)(ap + 16);
      if (t == 0) __builtin_prefetch(ap + 32, 0, 1);   // global_prefetch_b8
#pragma unroll
      for (int i = 0; i < 8; ++i) { a[t][i] = lo[i]; a[t][i + 8] = hi[i]; }
    }
  };

  constexpr int NK = K / 32;
  v8f  acc[MT] = {};
  v16h aA[MT], aB[MT];
  v16h bA, bB;

  loadB(0, bA);
  loadA(0, aA);
#pragma unroll
  for (int kk = 0; kk < NK; ++kk) {
    const bool even = (kk & 1) == 0;
    v16h* aCur = even ? aA : aB;
    v16h* aNxt = even ? aB : aA;
    v16h& bCur = even ? bA : bB;
    v16h& bNxt = even ? bB : bA;
    if (kk + 1 < NK) {                       // issue next k-step's loads first
      loadB((kk + 1) * 32, bNxt);
      loadA((kk + 1) * 32, aNxt);
    }
#pragma unroll
    for (int t = 0; t < MT; ++t) {
      // 8 args: (neg_a, A, neg_b, B, c_mod, C, reuse_a, reuse_b)
      acc[t] = __builtin_amdgcn_wmma_f32_16x16x32_f16(false, aCur[t], false, bCur,
                                                      (short)0, acc[t], false, false);
    }
  }

  // C/D layout: VGPR i -> (lanes 0-15: M=i, N=lane) (lanes 16-31: M=i+8, N=lane-16)
  const int ccol = n0 + l15;
  if (ccol < ldC) {                        // only ragged for the padded 40-class head
    float bv = bias ? bias[ccol] : 0.0f;
#pragma unroll
    for (int t = 0; t < MT; ++t) {
      int rbase = m0 + t * 16 + (half << 3);
#pragma unroll
      for (int i = 0; i < 8; ++i)
        C[(size_t)(rbase + i) * ldC + ccol] = acc[t][i] + bv;
    }
  }
}

// ---------------- per-node attention scalars: al/ar[n,h] = <H[n,h,:], att[h,:]> ----------------
__global__ void alr_kernel(const float* __restrict__ H, const float* __restrict__ attl,
                           const float* __restrict__ attr, float* __restrict__ al,
                           float* __restrict__ ar) {
  int t = blockIdx.x * blockDim.x + threadIdx.x;     // n*HEADS + h
  if (t >= N_NODES * HEADS) return;
  int n = t / HEADS, h = t % HEADS;
  const float* hp = H + (size_t)n * HC + h * HID;
  const float* lp = attl + h * HID;
  const float* rp = attr + h * HID;
  float sl = 0.f, sr = 0.f;
#pragma unroll 8
  for (int c = 0; c < HID; ++c) { float v = hp[c]; sl += v * lp[c]; sr += v * rp[c]; }
  al[t] = sl; ar[t] = sr;
}

__device__ __forceinline__ void edge_sd(const long long* ei, int e, int& s, int& d) {
  if (e < N_EDGES) { s = (int)ei[e]; d = (int)ei[N_EDGES + e]; }
  else             { s = e - N_EDGES; d = s; }       // appended self loops
}

// ---------------- alpha = leaky_relu(al[src]+ar[dst]); segment max via atomic on encoded uint ----
__global__ void edge_alpha_max(const long long* __restrict__ ei,
                               const float* __restrict__ al, const float* __restrict__ ar,
                               float* __restrict__ alpha, unsigned* __restrict__ menc) {
  int e = blockIdx.x * blockDim.x + threadIdx.x;
  if (e >= ET) return;
  int s, d; edge_sd(ei, e, s, d);
#pragma unroll
  for (int h = 0; h < HEADS; ++h) {
    float v = al[s * HEADS + h] + ar[d * HEADS + h];
    v = v > 0.f ? v : NEG_SLOPE * v;
    alpha[(size_t)e * HEADS + h] = v;
    atomicMax(&menc[d * HEADS + h], fenc(v));
  }
}

// ---------------- e = exp(alpha - m[dst]); segment sum ----------------
__global__ void edge_exp_sum(const long long* __restrict__ ei,
                             float* __restrict__ alpha, const unsigned* __restrict__ menc,
                             float* __restrict__ ssum) {
  int e = blockIdx.x * blockDim.x + threadIdx.x;
  if (e >= ET) return;
  int s, d; edge_sd(ei, e, s, d); (void)s;
#pragma unroll
  for (int h = 0; h < HEADS; ++h) {
    size_t idx = (size_t)e * HEADS + h;
    float m  = fdec(menc[d * HEADS + h]);            // every dst has a self loop -> valid
    float ex = __expf(alpha[idx] - m);
    alpha[idx] = ex;                                 // reuse buffer for e
    atomicAdd(&ssum[d * HEADS + h], ex);
  }
}

// ---------------- message scatter: one wave32 per edge, 6 channels/lane ----------------
__global__ __launch_bounds__(256)
void edge_scatter(const long long* __restrict__ ei, const float* __restrict__ H,
                  const float* __restrict__ eexp, const float* __restrict__ ssum,
                  float* __restrict__ agg) {
  int e    = blockIdx.x * (blockDim.x >> 5) + (threadIdx.x >> 5);
  int lane = threadIdx.x & 31;
  if (e >= ET) return;
  int s, d; edge_sd(ei, e, s, d);
  const float* hs = H   + (size_t)s * HC;
  float*       ad = agg + (size_t)d * HC;
#pragma unroll
  for (int h = 0; h < HEADS; ++h) {
    float a = eexp[(size_t)e * HEADS + h] / (ssum[d * HEADS + h] + 1e-16f);
    int c0 = h * HID + lane;                         // lanes cover 64 channels in 2 steps
    atomicAdd(&ad[c0],      hs[c0]      * a);
    atomicAdd(&ad[c0 + 32], hs[c0 + 32] * a);
  }
}

// ---------------- bias + ReLU + convert to f16 for next GEMM (4-wide, 4 | 192) --------------
__global__ void bias_relu_cvt(const float4* __restrict__ agg, const float* __restrict__ b,
                              v4h* __restrict__ out) {
  int i = blockIdx.x * blockDim.x + threadIdx.x;     // over (N*HC)/4
  if (i >= N_NODES * HC / 4) return;
  int c = (i * 4) % HC;
  float4 v = agg[i];
  v4h o;
  float vx = v.x + b[c + 0]; o[0] = (_Float16)(vx > 0.f ? vx : 0.f);
  float vy = v.y + b[c + 1]; o[1] = (_Float16)(vy > 0.f ? vy : 0.f);
  float vz = v.z + b[c + 2]; o[2] = (_Float16)(vz > 0.f ? vz : 0.f);
  float vw = v.w + b[c + 3]; o[3] = (_Float16)(vw > 0.f ? vw : 0.f);
  out[i] = o;
}

// ---------------- host side ----------------
static inline size_t align256(size_t x) { return (x + 255u) & ~(size_t)255u; }

extern "C" void kernel_launch(void* const* d_in, const int* in_sizes, int n_in,
                              void* d_out, int out_size, void* d_ws, size_t ws_size,
                              hipStream_t stream) {
  (void)in_sizes; (void)n_in; (void)out_size; (void)ws_size;
  const float*      x   = (const float*)d_in[0];
  const long long*  ei  = (const long long*)d_in[1];   // int64 [2,E]
  const float* W1   = (const float*)d_in[2];
  const float* atl1 = (const float*)d_in[3];
  const float* atr1 = (const float*)d_in[4];
  const float* b1   = (const float*)d_in[5];
  const float* W2   = (const float*)d_in[6];
  const float* atl2 = (const float*)d_in[7];
  const float* atr2 = (const float*)d_in[8];
  const float* b2   = (const float*)d_in[9];
  const float* Wo   = (const float*)d_in[10];
  const float* bo   = (const float*)d_in[11];
  float* out = (float*)d_out;

  // workspace carve-up (~117 MB)
  char* ws = (char*)d_ws;
  size_t o = 0;
  auto take = [&](size_t bytes) { size_t r = o; o = align256(o + bytes); return r; };
  _Float16* xbf  = (_Float16*)(ws + take((size_t)N_NODES * IN_DIM * 2));
  _Float16* w1bf = (_Float16*)(ws + take((size_t)HC * IN_DIM * 2));
  _Float16* w2bf = (_Float16*)(ws + take((size_t)HC * HC * 2));
  _Float16* wobf = (_Float16*)(ws + take((size_t)NCLS_PAD * HC * 2));  // zero-padded 48x192
  float*    H    = (float*)   (ws + take((size_t)N_NODES * HC * 4));
  float*    al   = (float*)   (ws + take((size_t)N_NODES * HEADS * 4));
  float*    ar   = (float*)   (ws + take((size_t)N_NODES * HEADS * 4));
  unsigned* menc = (unsigned*)(ws + take((size_t)N_NODES * HEADS * 4));
  float*    ssum = (float*)   (ws + take((size_t)N_NODES * HEADS * 4));
  float*    ealp = (float*)   (ws + take((size_t)ET * HEADS * 4));
  float*    agg  = (float*)   (ws + take((size_t)N_NODES * HC * 4));
  _Float16* hbf  = (_Float16*)(ws + take((size_t)N_NODES * HC * 2));

  const int TB = 256;
  auto blocks = [](long long n, int tb) { return (unsigned)((n + tb - 1) / tb); };

  // weight/feature converts (all counts are multiples of 4)
  cvt_f32_f16x4<<<blocks((long long)N_NODES * IN_DIM / 4, TB), TB, 0, stream>>>(
      (const float4*)x, (v4h*)xbf, N_NODES * IN_DIM / 4);
  cvt_f32_f16x4<<<blocks(HC * IN_DIM / 4, TB), TB, 0, stream>>>(
      (const float4*)W1, (v4h*)w1bf, HC * IN_DIM / 4);
  cvt_f32_f16x4<<<blocks(HC * HC / 4, TB), TB, 0, stream>>>(
      (const float4*)W2, (v4h*)w2bf, HC * HC / 4);
  hipMemsetAsync(wobf, 0, (size_t)NCLS_PAD * HC * 2, stream);     // zero rows 40..47
  cvt_f32_f16x4<<<blocks(NCLS * HC / 4, TB), TB, 0, stream>>>(
      (const float4*)Wo, (v4h*)wobf, NCLS * HC / 4);

  const dim3 gblk(128);
  const unsigned gx = N_NODES / (16 * MT);            // 625, exact

  auto edge_phase = [&](const float* atl, const float* atr, const float* bias) {
    alr_kernel<<<blocks((long long)N_NODES * HEADS, TB), TB, 0, stream>>>(H, atl, atr, al, ar);
    hipMemsetAsync(menc, 0, (size_t)N_NODES * HEADS * 4, stream);
    hipMemsetAsync(ssum, 0, (size_t)N_NODES * HEADS * 4, stream);
    hipMemsetAsync(agg,  0, (size_t)N_NODES * HC * 4, stream);
    edge_alpha_max<<<blocks(ET, TB), TB, 0, stream>>>(ei, al, ar, ealp, menc);
    edge_exp_sum<<<blocks(ET, TB), TB, 0, stream>>>(ei, ealp, menc, ssum);
    edge_scatter<<<blocks((long long)ET * 32, TB), TB, 0, stream>>>(ei, H, ealp, ssum, agg);
    bias_relu_cvt<<<blocks((long long)N_NODES * HC / 4, TB), TB, 0, stream>>>(
        (const float4*)agg, bias, (v4h*)hbf);
  };

  // layer 1: x[.,128] -> hbf[.,192]
  gemm_wmma_f16<IN_DIM><<<dim3(gx, HC / 64), gblk, 0, stream>>>(xbf, w1bf, nullptr, H, HC, HC);
  edge_phase(atl1, atr1, b1);
  // layer 2: hbf[.,192] -> hbf[.,192]
  gemm_wmma_f16<HC><<<dim3(gx, HC / 64), gblk, 0, stream>>>(hbf, w2bf, nullptr, H, HC, HC);
  edge_phase(atl2, atr2, b2);
  // output head: out[.,40] = hbf @ Wo(padded 48x192)^T + bo  (store guard at col<40)
  gemm_wmma_f16<HC><<<dim3(gx, 1), gblk, 0, stream>>>(hbf, wobf, bo, out, NCLS_PAD, NCLS);
}
